// DenseRNNBase_89266600280788
// MI455X (gfx1250) — compile-verified, matrix operand order checked by reading, not run on hardware
//
#include <hip/hip_runtime.h>

#define TT 256
#define BB 128
#define II 512
#define HH 512
#define N3 1536
#define NTILES 96          // N3 / 16
#define NBLK 96            // persistent grid: one 16-col N-tile per block
#define BH (BB * HH)

typedef __attribute__((ext_vector_type(16))) __bf16 v16bf;
typedef __attribute__((ext_vector_type(8)))  __bf16 v8bf;
typedef __attribute__((ext_vector_type(8)))  float  v8f;

__device__ __forceinline__ __bf16 f2bf(float f) {
  union { float f; unsigned u; } v; v.f = f;
  unsigned r = v.u + 0x7FFFu + ((v.u >> 16) & 1u);   // round-to-nearest-even
  union { unsigned short s; __bf16 b; } o; o.s = (unsigned short)(r >> 16);
  return o.b;
}

// ---------------- one-time setup kernels ----------------

__global__ void k_f32_to_bf16(const float* __restrict__ src, __bf16* __restrict__ dst, int n) {
  int i = blockIdx.x * blockDim.x + threadIdx.x;
  if (i < n) dst[i] = f2bf(src[i]);
}

// Pre-swizzle stacked weights [W_ih; W_hh; W_dense[0..6]] into WMMA B-fragment order:
// dst[((c*96 + nt)*16 + kc)*512 + lane*16 + elem] -> each lane's v16bf is one
// contiguous 32B run matching the ISA 16-bit B layout (n = lane&15, kh = 8*(lane>>4),
// K runs kh..kh+7 and 16+kh..16+kh+7 within the 32-deep k-chunk).
__global__ void k_swizzle_w(const float* __restrict__ Wih, const float* __restrict__ Whh,
                            const float* __restrict__ Wd, __bf16* __restrict__ dst, int n) {
  int idx = blockIdx.x * blockDim.x + threadIdx.x;
  if (idx >= n) return;
  int i  = idx & 15;            // element within lane fragment
  int l  = (idx >> 4) & 31;     // lane
  int kc = (idx >> 9) & 15;     // k-chunk of 32
  int nt = (idx >> 13) % NTILES;
  int c  = (idx >> 13) / NTILES;
  int kh  = (l >> 4) * 8;
  int kof = (i < 8) ? (kh + i) : (16 + kh + (i - 8));
  int k   = kc * 32 + kof;
  int nn  = nt * 16 + (l & 15);
  const int S = HH * N3;
  size_t si = (size_t)k * N3 + nn;
  float v = (c == 0) ? Wih[si] : (c == 1) ? Whh[si] : Wd[(size_t)(c - 2) * S + si];
  dst[idx] = f2bf(v);
}

// bf16 ring of 8 hidden states; h[tau] lives in slot tau&7; h0 == h[-1] -> slot 7.
// Also zero-initializes the software grid-barrier cells (determinism per launch).
__global__ void k_init_ring(const float* __restrict__ h0, __bf16* __restrict__ ring,
                            unsigned* __restrict__ cnt, unsigned* __restrict__ gen, int n) {
  int i = blockIdx.x * blockDim.x + threadIdx.x;
  if (i == 0) { cnt[0] = 0u; gen[0] = 0u; }
  if (i >= n) return;
  const int S = BH;
  int slot = i / S, rem = i % S;
  ring[i] = (slot == 7) ? f2bf(h0[rem]) : f2bf(0.0f);
}

// ---------------- persistent RNN kernel ----------------

__device__ __forceinline__ void grid_barrier(unsigned* cnt, unsigned* gen) {
  __syncthreads();
  if (threadIdx.x == 0) {
    __threadfence();   // make Sx/Sh/ring/out stores device-visible
    unsigned g = __hip_atomic_load(gen, __ATOMIC_ACQUIRE, __HIP_MEMORY_SCOPE_AGENT);
    unsigned prev = __hip_atomic_fetch_add(cnt, 1u, __ATOMIC_ACQ_REL, __HIP_MEMORY_SCOPE_AGENT);
    if (prev == NBLK - 1) {
      __hip_atomic_store(cnt, 0u, __ATOMIC_RELAXED, __HIP_MEMORY_SCOPE_AGENT);
      __hip_atomic_fetch_add(gen, 1u, __ATOMIC_RELEASE, __HIP_MEMORY_SCOPE_AGENT);
    } else {
      while (__hip_atomic_load(gen, __ATOMIC_ACQUIRE, __HIP_MEMORY_SCOPE_AGENT) == g)
        __builtin_amdgcn_s_sleep(2);
    }
  }
  __syncthreads();
}

__device__ __forceinline__ v16bf load_afrag(const __bf16* ap) {
  // 16-bit A layout: two contiguous 16B runs per lane
  v8bf lo = *reinterpret_cast<const v8bf*>(ap);
  v8bf hi = *reinterpret_cast<const v8bf*>(ap + 16);
  v16bf f;
#pragma unroll
  for (int i = 0; i < 8; ++i) { f[i] = lo[i]; f[i + 8] = hi[i]; }
  return f;
}

__device__ __forceinline__ v8f wmma_bf16(v16bf a, v16bf b, v8f c) {
  return __builtin_amdgcn_wmma_f32_16x16x32_bf16(false, a, false, b, (short)0, c, false, false);
}

__global__ __launch_bounds__(256) void k_rnn(
    const __bf16* __restrict__ xbf, const __bf16* __restrict__ wsw,
    __bf16* __restrict__ ring, float* __restrict__ Sx, float* __restrict__ Sh,
    float* __restrict__ out, const float* __restrict__ h0,
    const float* __restrict__ bih, const float* __restrict__ bhh,
    unsigned* cnt, unsigned* gen) {
  const int wave = threadIdx.x >> 5;       // M tile 0..7
  const int lane = threadIdx.x & 31;
  const int m0   = wave * 16;
  const int mrow = lane & 15;
  const int kh   = (lane >> 4) * 8;
  const int nt   = blockIdx.x;             // this block's N tile
  const int gtid = blockIdx.x * 256 + threadIdx.x;

  for (int t = 0; t < TT; ++t) {
    const int nmat = 2 + (t & 7);          // W_ih + W_hh + (t%8) dense slices
    // even/odd-K split accumulators: two independent WMMA chains per wave
    v8f ax0 = {}, ax1 = {}, ah0 = {}, ah1 = {};

    // ---- chunk 0: x_t @ W_ih (kept separate: gates need i_n alone) ----
    {
      const __bf16* A = xbf + (size_t)t * BB * II + (size_t)(m0 + mrow) * II + kh;
      const __bf16* B = wsw + ((size_t)nt * 16) * 512 + lane * 16;
#pragma unroll
      for (int kc = 0; kc < 16; kc += 2) {
        v16bf a0 = load_afrag(A + kc * 32);
        v16bf a1 = load_afrag(A + kc * 32 + 32);
        v16bf b0 = *reinterpret_cast<const v16bf*>(B + (size_t)kc * 512);
        v16bf b1 = *reinterpret_cast<const v16bf*>(B + (size_t)(kc + 1) * 512);
        ax0 = wmma_bf16(a0, b0, ax0);
        ax1 = wmma_bf16(a1, b1, ax1);
      }
    }
    // ---- chunks 1..nmat-1: h[t-1]@W_hh + dense skips ----
    for (int c = 1; c < nmat; ++c) {
      int slot = (t - 1 - (c >= 2 ? (c - 2) : 0)) & 7;
      const __bf16* A = ring + (size_t)slot * BH + (size_t)(m0 + mrow) * HH + kh;
      const __bf16* B = wsw + ((size_t)(c * NTILES + nt) * 16) * 512 + lane * 16;
#pragma unroll
      for (int kc = 0; kc < 16; kc += 2) {
        v16bf a0 = load_afrag(A + kc * 32);
        v16bf a1 = load_afrag(A + kc * 32 + 32);
        v16bf b0 = *reinterpret_cast<const v16bf*>(B + (size_t)kc * 512);
        v16bf b1 = *reinterpret_cast<const v16bf*>(B + (size_t)(kc + 1) * 512);
        ah0 = wmma_bf16(a0, b0, ah0);
        ah1 = wmma_bf16(a1, b1, ah1);
      }
    }
    // ---- write S tiles (C/D layout: m = j + 8*(lane>>4), n = lane&15) ----
#pragma unroll
    for (int j = 0; j < 8; ++j) {
      int m = m0 + j + 8 * (lane >> 4);
      int n = nt * 16 + (lane & 15);
      Sx[(size_t)m * N3 + n] = ax0[j] + ax1[j];
      Sh[(size_t)m * N3 + n] = ah0[j] + ah1[j];
    }

    grid_barrier(cnt, gen);   // S ready device-wide

    // ---- gates: grid-stride over B*H ----
    const float* hprev = (t == 0) ? h0 : out + (size_t)(t - 1) * BH;
    float* hout = out + (size_t)t * BH;
    __bf16* rslot = ring + (size_t)(t & 7) * BH;
    for (int i = gtid; i < BH; i += NBLK * 256) {
      int h = i & (HH - 1);
      size_t base = (size_t)(i >> 9) * N3 + h;
      float pre_r = Sx[base]        + bih[h]        + Sh[base]        + bhh[h];
      float pre_z = Sx[base + 512]  + bih[512 + h]  + Sh[base + 512]  + bhh[512 + h];
      float i_n   = Sx[base + 1024] + bih[1024 + h];
      float h_n   = Sh[base + 1024] + bhh[1024 + h];
      float r = 1.0f / (1.0f + __expf(-pre_r));
      float z = 1.0f / (1.0f + __expf(-pre_z));
      float n = tanhf(i_n + r * h_n);
      float hp = hprev[i];
      float hy = n + z * (hp - n);
      hout[i] = hy;
      rslot[i] = f2bf(hy);
    }

    grid_barrier(cnt, gen);   // protect WAR on Sx/Sh and ring reads next step
  }
}

// ---------------- launch ----------------

extern "C" void kernel_launch(void* const* d_in, const int* in_sizes, int n_in,
                              void* d_out, int out_size, void* d_ws, size_t ws_size,
                              hipStream_t stream) {
  (void)in_sizes; (void)n_in; (void)out_size; (void)ws_size;
  const float* x   = (const float*)d_in[0];
  const float* h0  = (const float*)d_in[1];
  const float* Wih = (const float*)d_in[2];
  const float* Whh = (const float*)d_in[3];
  const float* bih = (const float*)d_in[4];
  const float* bhh = (const float*)d_in[5];
  const float* Wd  = (const float*)d_in[6];
  float* out = (float*)d_out;

  char* ws = (char*)d_ws;
  size_t off = 0;
  auto alloc = [&](size_t bytes) {
    char* p = ws + off;
    off = (off + bytes + 255) & ~size_t(255);
    return p;
  };
  __bf16*   xbf  = (__bf16*)alloc((size_t)TT * BB * II * 2);   // 32 MB
  __bf16*   wsw  = (__bf16*)alloc((size_t)9 * HH * N3 * 2);    // 13.5 MB, L2-resident
  __bf16*   ring = (__bf16*)alloc((size_t)8 * BH * 2);         // 1 MB
  float*    Sx   = (float*)alloc((size_t)BB * N3 * 4);
  float*    Sh   = (float*)alloc((size_t)BB * N3 * 4);
  unsigned* cnt  = (unsigned*)alloc(256);
  unsigned* gen  = (unsigned*)alloc(256);

  { int n = TT * BB * II; k_f32_to_bf16<<<(n + 255) / 256, 256, 0, stream>>>(x, xbf, n); }
  { int n = 9 * HH * N3;  k_swizzle_w<<<(n + 255) / 256, 256, 0, stream>>>(Wih, Whh, Wd, wsw, n); }
  { int n = 8 * BH;       k_init_ring<<<(n + 255) / 256, 256, 0, stream>>>(h0, ring, cnt, gen, n); }

  k_rnn<<<NBLK, 256, 0, stream>>>(xbf, wsw, ring, Sx, Sh, out, h0, bih, bhh, cnt, gen);
}
